// Attention_4758823764404
// MI455X (gfx1250) — matrix operand order repaired and checked by reading, not compiled, not run on hardware
//
#include <hip/hip_runtime.h>

// ---------------------------------------------------------------------------
// MI455X (gfx1250) multi-head attention, f16 WMMA + fp32 accum.
// B=4, S=2048, D=1024, H=16, hd=64, scale=8.
// Data movement: TDM tensor_load_to_lds (6-arg builtin) or b128 staging.
// ---------------------------------------------------------------------------

typedef __attribute__((ext_vector_type(16))) _Float16 v16h;
typedef __attribute__((ext_vector_type(8)))  _Float16 v8h;
typedef __attribute__((ext_vector_type(4)))  _Float16 v4h;
typedef __attribute__((ext_vector_type(8)))  float    v8f;
typedef __attribute__((ext_vector_type(4)))  float    v4f;
typedef __attribute__((ext_vector_type(4)))  unsigned int v4u;
typedef __attribute__((ext_vector_type(8)))  int      v8i;
typedef __attribute__((ext_vector_type(4)))  int      v4i;

#define DIMS   1024
#define SEQ    2048
#define BATCH  4
#define NHEAD  16
#define HD     64
#define MTOT   (BATCH * SEQ)
#define TK     32
#define KB     32

#if __has_builtin(__builtin_amdgcn_tensor_load_to_lds) && __has_builtin(__builtin_amdgcn_s_wait_tensorcnt)
#define USE_TDM 1
#else
#define USE_TDM 0
#endif

__device__ __forceinline__ v16h cat8(v8h lo, v8h hi) {
    return __builtin_shufflevector(lo, hi, 0,1,2,3,4,5,6,7,8,9,10,11,12,13,14,15);
}

#if USE_TDM
// Build the loop-invariant D# group1 (2-byte elements, LDS padding enabled).
// pad_interval code: 3 -> every 16 dwords, 4 -> every 32 dwords; pad_amount 3 -> 4 dwords.
__device__ __forceinline__ v8i tdm_g1(unsigned dim0, unsigned dim1,
                                      unsigned tile0, unsigned tile1,
                                      unsigned stride0,
                                      unsigned pad_i, unsigned pad_a)
{
    v8i g1;
    g1[0] = (int)((1u << 16) | (1u << 20) | (pad_i << 22) | (pad_a << 25));
    g1[1] = (int)((dim0 & 0xFFFFu) << 16);
    g1[2] = (int)(((dim0 >> 16) & 0xFFFFu) | ((dim1 & 0xFFFFu) << 16));
    g1[3] = (int)(((dim1 >> 16) & 0xFFFFu) | (tile0 << 16));
    g1[4] = (int)(tile1 & 0xFFFFu);
    g1[5] = (int)stride0;
    g1[6] = 0;
    g1[7] = 0;
    return g1;
}

__device__ __forceinline__ void tdm_issue(unsigned lds_addr, const void* gaddr, v8i g1)
{
    unsigned long long ga = (unsigned long long)(size_t)gaddr;
    v4u g0;
    g0[0] = 1u;                                                 // count=1
    g0[1] = lds_addr;                                           // LDS byte addr
    g0[2] = (unsigned)(ga & 0xFFFFFFFFu);
    g0[3] = (unsigned)((ga >> 32) & 0x01FFFFFFu) | (2u << 30);  // hi | type=2
    v4i gz4 = {0, 0, 0, 0};
    v8i gz8 = {0, 0, 0, 0, 0, 0, 0, 0};
    __builtin_amdgcn_tensor_load_to_lds(g0, g1, gz4, gz4, gz8, 0);
}
#endif

// ---------------------------------------------------------------------------
__global__ void f32_to_f16_kernel(const v4f* __restrict__ in,
                                  v4h* __restrict__ out, int n4) {
    int i = blockIdx.x * blockDim.x + threadIdx.x;
    if (i < n4) {
        v4f v = in[i];
        v4h h;
        for (int j = 0; j < 4; ++j) h[j] = (_Float16)v[j];
        out[i] = h;
    }
}

// ---------------------------------------------------------------------------
// GEMM: out = A(f16 [M,K]) @ B(f32 [K,N] -> f16) + bias.  64x64 block tile.
// ---------------------------------------------------------------------------
__global__ __launch_bounds__(128)
void gemm_wmma_kernel(const _Float16* __restrict__ A,
                      const float* __restrict__ Bw,
                      const float* __restrict__ bias,
                      _Float16* __restrict__ outH,
                      float* __restrict__ outF,
                      int M, int N, int K)
{
    __shared__ _Float16 As[64][TK + 8];     // [m][k]
    __shared__ _Float16 Bst[64][TK + 8];    // transposed: [n][k]

    const int tid  = threadIdx.x;
    const int lane = tid & 31;
    const int wv   = tid >> 5;
    const int mrow = lane & 15;
    const int hg   = lane >> 4;
    const int n0   = blockIdx.x * 64;
    const int m0   = blockIdx.y * 64;

    v8f acc[4];
    for (int t = 0; t < 4; ++t)
        for (int r = 0; r < 8; ++r) acc[t][r] = 0.0f;

    // B staging: uniform base pointer (SALU-advanced) + 32-bit thread offsets
    // so the loads lower to global_load_b128 saddr+voffset (GVS form).
    const int bc4 = tid & 15;
    const int br  = tid >> 4;
    unsigned boff[4];
    for (int i = 0; i < 4; ++i)
        boff[i] = (unsigned)((br + i * 8) * N + n0 + bc4 * 4);
    const float* Bk = Bw;                       // advances TK*N per k-step

#if USE_TDM
    const v8i g1A = tdm_g1(TK, 64, TK, 64, (unsigned)K, /*16dw*/3, /*4dw*/3);
    const _Float16* ag = A + (size_t)m0 * K;
#else
    const int ar = tid >> 2;
    const int ac = tid & 3;
    const uint4* ap0 = (const uint4*)(A + (size_t)(m0 + ar) * K) + ac;
    const uint4* ap1 = (const uint4*)(A + (size_t)(m0 + 32 + ar) * K) + ac;
#endif

    for (int k0 = 0; k0 < K; k0 += TK) {
#if USE_TDM
        if (wv == 0)
            tdm_issue((unsigned)(size_t)&As[0][0], ag + k0, g1A);
#else
        *(uint4*)&As[ar][ac * 8]      = ap0[k0 / 8];
        *(uint4*)&As[32 + ar][ac * 8] = ap1[k0 / 8];
#endif
        // B tile: issue all loads, then convert + transposed store
        v4f bv4[4];
        for (int i = 0; i < 4; ++i) bv4[i] = *(const v4f*)(Bk + boff[i]);
        Bk += (size_t)TK * N;
        for (int i = 0; i < 4; ++i) {
            const int r = br + i * 8;
            Bst[bc4 * 4 + 0][r] = (_Float16)bv4[i][0];
            Bst[bc4 * 4 + 1][r] = (_Float16)bv4[i][1];
            Bst[bc4 * 4 + 2][r] = (_Float16)bv4[i][2];
            Bst[bc4 * 4 + 3][r] = (_Float16)bv4[i][3];
        }
#if USE_TDM
        if (wv == 0) __builtin_amdgcn_s_wait_tensorcnt(0);
#endif
        __syncthreads();

        const int arow = wv * 16 + mrow;
        v16h a = cat8(*(const v8h*)&As[arow][hg * 8],
                      *(const v8h*)&As[arow][16 + hg * 8]);
        for (int nt = 0; nt < 4; ++nt) {
            const int col = nt * 16 + mrow;
            v16h b = cat8(*(const v8h*)&Bst[col][hg * 8],
                          *(const v8h*)&Bst[col][16 + hg * 8]);
            acc[nt] = __builtin_amdgcn_wmma_f32_16x16x32_f16(
                false, a, false, b, (short)0, acc[nt], false, false);
        }
        __syncthreads();
    }

    for (int nt = 0; nt < 4; ++nt) {
        int col = n0 + nt * 16 + mrow;
        float bv = bias ? bias[col] : 0.0f;
        for (int r = 0; r < 8; ++r) {
            int row = m0 + wv * 16 + r + hg * 8;
            float v = acc[nt][r] + bv;
            if (outH) outH[(size_t)row * N + col] = (_Float16)v;
            else      outF[(size_t)row * N + col] = v;
        }
    }
}

// ---------------------------------------------------------------------------
// Flash attention.  Block = (b, h, 64 q-rows), 4 waves.  K/V 32x64 tiles via
// TDM; V transposed once per tile in LDS so all fragments are b128 loads.
// ---------------------------------------------------------------------------
__global__ __launch_bounds__(128)
void attn_wmma_kernel(const _Float16* __restrict__ qkv,   // [M, 3D] f16
                      _Float16* __restrict__ attnOut)     // [M, D]  f16
{
    __shared__ _Float16 Ks[KB][HD + 8];          // [key][d]
    __shared__ _Float16 Vs[KB][HD + 8];          // [key][d]
    __shared__ _Float16 Vt[HD][KB + 8];          // [d][key]
    __shared__ _Float16 Ps[4][16][KB + 8];

    const int tid  = threadIdx.x;
    const int lane = tid & 31;
    const int wv   = tid >> 5;
    const int mrow = lane & 15;
    const int hg   = lane >> 4;

    const int bh = blockIdx.y;
    const int b  = bh >> 4;
    const int h  = bh & 15;
    const int qb = blockIdx.x * 64 + wv * 16;

    const size_t rowStride = 3 * DIMS;
    const size_t kOff = DIMS     + (size_t)h * HD;
    const size_t vOff = 2 * DIMS + (size_t)h * HD;

    // Q fragments for the whole block
    v16h aq[2];
    {
        const _Float16* qrow =
            qkv + (size_t)(b * SEQ + qb + mrow) * rowStride + (size_t)h * HD;
        aq[0] = cat8(*(const v8h*)(qrow + hg * 8),
                     *(const v8h*)(qrow + 16 + hg * 8));
        aq[1] = cat8(*(const v8h*)(qrow + 32 + hg * 8),
                     *(const v8h*)(qrow + 48 + hg * 8));
    }

    float mrun[8], lrun[8];
    v8f o[4];
    for (int r = 0; r < 8; ++r) { mrun[r] = -1e30f; lrun[r] = 0.0f; }
    for (int nt = 0; nt < 4; ++nt)
        for (int r = 0; r < 8; ++r) o[nt][r] = 0.0f;

    const float c_exp = 1.4426950408889634f / 8.0f;   // log2(e)/sqrt(hd)

#if USE_TDM
    const v8i g1KV = tdm_g1(HD, KB, HD, KB, (unsigned)rowStride, /*32dw*/4, /*4dw*/3);
    const _Float16* kg = qkv + (size_t)(b * SEQ) * rowStride + kOff;
    const _Float16* vg = qkv + (size_t)(b * SEQ) * rowStride + vOff;
    const size_t tstep = (size_t)KB * rowStride;
#else
    const int sr = tid >> 3;
    const int sc = tid & 7;
    const uint4* kp0 = (const uint4*)(qkv + (size_t)(b * SEQ + sr) * rowStride + kOff) + sc;
    const uint4* kp1 = (const uint4*)(qkv + (size_t)(b * SEQ + 16 + sr) * rowStride + kOff) + sc;
    const uint4* vp0 = (const uint4*)(qkv + (size_t)(b * SEQ + sr) * rowStride + vOff) + sc;
    const uint4* vp1 = (const uint4*)(qkv + (size_t)(b * SEQ + 16 + sr) * rowStride + vOff) + sc;
    const size_t stepU4 = KB * rowStride / 8;
#endif

    // transpose work split: thread handles key-row tr, d range [td, td+16)
    const int tr = tid >> 2;
    const int td = (tid & 3) * 16;

    for (int kb = 0; kb < SEQ; kb += KB) {
#if USE_TDM
        if (wv == 0) {
            tdm_issue((unsigned)(size_t)&Ks[0][0], kg, g1KV);
            tdm_issue((unsigned)(size_t)&Vs[0][0], vg, g1KV);
            __builtin_amdgcn_s_wait_tensorcnt(0);
        }
        kg += tstep; vg += tstep;
#else
        *(uint4*)&Ks[sr][sc * 8]      = *kp0;
        *(uint4*)&Ks[16 + sr][sc * 8] = *kp1;
        *(uint4*)&Vs[sr][sc * 8]      = *vp0;
        *(uint4*)&Vs[16 + sr][sc * 8] = *vp1;
        kp0 += stepU4; kp1 += stepU4; vp0 += stepU4; vp1 += stepU4;
#endif
        __syncthreads();

        // cooperative V transpose: Vs[key][d] -> Vt[d][key]
        {
            v8h va = *(const v8h*)&Vs[tr][td];
            v8h vb = *(const v8h*)&Vs[tr][td + 8];
            for (int i = 0; i < 8; ++i) Vt[td + i][tr]     = va[i];
            for (int i = 0; i < 8; ++i) Vt[td + 8 + i][tr] = vb[i];
        }
        __syncthreads();

        // scores S = Q K^T : two 16-key tiles, hd reduced in 2 chunks of 32
        v8f s[2];
        for (int t = 0; t < 2; ++t)
            for (int r = 0; r < 8; ++r) s[t][r] = 0.0f;
        for (int t = 0; t < 2; ++t)
            for (int cc = 0; cc < 2; ++cc) {
                const _Float16* krow = &Ks[t * 16 + mrow][cc * 32 + hg * 8];
                v16h bk = cat8(*(const v8h*)krow, *(const v8h*)(krow + 16));
                s[t] = __builtin_amdgcn_wmma_f32_16x16x32_f16(
                    false, aq[cc], false, bk, (short)0, s[t], false, false);
            }

        // online softmax (row lives across the 16-lane C-layout half)
        float mnew[8], scl[8];
        for (int r = 0; r < 8; ++r) {
            float v = fmaxf(s[0][r], s[1][r]);
            v = fmaxf(v, __shfl_xor(v, 1, 32));
            v = fmaxf(v, __shfl_xor(v, 2, 32));
            v = fmaxf(v, __shfl_xor(v, 4, 32));
            v = fmaxf(v, __shfl_xor(v, 8, 32));
            float mn = fmaxf(mrun[r], v);
            scl[r]  = __builtin_exp2f((mrun[r] - mn) * c_exp);
            mnew[r] = mn;
        }
        for (int r = 0; r < 8; ++r) {
            float p0 = __builtin_exp2f((s[0][r] - mnew[r]) * c_exp);
            float p1 = __builtin_exp2f((s[1][r] - mnew[r]) * c_exp);
            s[0][r] = p0; s[1][r] = p1;
            float t = p0 + p1;
            t += __shfl_xor(t, 1, 32);
            t += __shfl_xor(t, 2, 32);
            t += __shfl_xor(t, 4, 32);
            t += __shfl_xor(t, 8, 32);
            lrun[r] = lrun[r] * scl[r] + t;
            mrun[r] = mnew[r];
        }
        for (int nt = 0; nt < 4; ++nt)
            for (int r = 0; r < 8; ++r) o[nt][r] *= scl[r];

        // P: C layout -> per-wave LDS -> A layout (in-order DS within wave)
        for (int t = 0; t < 2; ++t)
            for (int r = 0; r < 8; ++r)
                Ps[wv][r + hg * 8][t * 16 + mrow] = (_Float16)s[t][r];
        __builtin_amdgcn_wave_barrier();
        v16h pa = cat8(*(const v8h*)&Ps[wv][mrow][hg * 8],
                       *(const v8h*)&Ps[wv][mrow][16 + hg * 8]);
        __builtin_amdgcn_wave_barrier();

        // O += P @ V : V fragments contiguous in Vt
        for (int nt = 0; nt < 4; ++nt) {
            const _Float16* vrow = &Vt[nt * 16 + mrow][hg * 8];
            v16h bv = cat8(*(const v8h*)vrow, *(const v8h*)(vrow + 16));
            o[nt] = __builtin_amdgcn_wmma_f32_16x16x32_f16(
                false, pa, false, bv, (short)0, o[nt], false, false);
        }
        __syncthreads();
    }

    for (int nt = 0; nt < 4; ++nt)
        for (int r = 0; r < 8; ++r) {
            int row = qb + r + hg * 8;
            int d   = nt * 16 + mrow;
            float v = o[nt][r] / lrun[r];
            attnOut[(size_t)(b * SEQ + row) * DIMS + h * HD + d] = (_Float16)v;
        }
}

// ---------------------------------------------------------------------------
extern "C" void kernel_launch(void* const* d_in, const int* in_sizes, int n_in,
                              void* d_out, int out_size, void* d_ws, size_t ws_size,
                              hipStream_t stream) {
    const float* x     = (const float*)d_in[0];
    const float* Wqkv  = (const float*)d_in[1];
    const float* bqkv  = (const float*)d_in[2];
    const float* Wproj = (const float*)d_in[3];
    const float* bproj = (const float*)d_in[4];
    float* out = (float*)d_out;

    char* ws = (char*)d_ws;
    _Float16* xh    = (_Float16*)(ws);                                   // 16 MiB
    _Float16* qkvh  = (_Float16*)(ws + (size_t)16 * 1024 * 1024);        // 48 MiB
    _Float16* attnh = (_Float16*)(ws + (size_t)64 * 1024 * 1024);        // 16 MiB

    f32_to_f16_kernel<<<(MTOT * DIMS / 4) / 256, 256, 0, stream>>>(
        (const v4f*)x, (v4h*)xh, MTOT * DIMS / 4);

    gemm_wmma_kernel<<<dim3((3 * DIMS) / 64, MTOT / 64), 128, 0, stream>>>(
        xh, Wqkv, bqkv, qkvh, nullptr, MTOT, 3 * DIMS, DIMS);

    attn_wmma_kernel<<<dim3(SEQ / 64, BATCH * NHEAD), 128, 0, stream>>>(qkvh, attnh);

    gemm_wmma_kernel<<<dim3(DIMS / 64, MTOT / 64), 128, 0, stream>>>(
        attnh, Wproj, bproj, nullptr, out, MTOT, DIMS, DIMS);
}
